// PointNetFeaturePropagation_23261542875829
// MI455X (gfx1250) — matrix operand order repaired and checked by reading, not compiled, not run on hardware
//
#include <hip/hip_runtime.h>
#include <stdint.h>

typedef __attribute__((ext_vector_type(16))) __bf16 v16bf;
typedef __attribute__((ext_vector_type(8)))  float  v8f;
typedef int vi4 __attribute__((vector_size(16)));   // matches async-LDS builtin param

#define B_   16
#define N_   4096
#define S_   1024
#define C1_  256
#define C2_  512
#define CIN_ 768
#define H_   256

#if defined(__has_builtin)
#if __has_builtin(__builtin_amdgcn_global_load_async_to_lds_b128)
#define HAVE_ASYNC_LDS 1
#endif
#if __has_builtin(__builtin_amdgcn_s_wait_asynccnt)
#define HAVE_WAIT_ASYNC_BUILTIN 1
#endif
#endif

static __device__ __forceinline__ void wait_async0() {
#if defined(HAVE_WAIT_ASYNC_BUILTIN)
    __builtin_amdgcn_s_wait_asynccnt(0);
#else
    asm volatile("s_wait_asynccnt 0x0" ::: "memory");
#endif
}

static __device__ __forceinline__ unsigned short f2bf(float f) {
    unsigned u = __float_as_uint(f);
    u += 0x7FFFu + ((u >> 16) & 1u);     // round-to-nearest-even
    return (unsigned short)(u >> 16);
}

// ---------------------------------------------------------------------------
// 1) KNN-3: one block per (batch, 256-query chunk); refs staged into LDS via
//    the CDNA5 async global->LDS path (ASYNCcnt) when available.
// ---------------------------------------------------------------------------
__global__ __launch_bounds__(256) void knn3_kernel(
        const float* __restrict__ xyz1, const float* __restrict__ xyz2,
        int* __restrict__ idx_out, float* __restrict__ w_out) {
    __shared__ float sx[S_], sy[S_], sz[S_];
    const int b = blockIdx.y;
    const int t = threadIdx.x;
    const float* ref = xyz2 + (size_t)b * 3 * S_;
#if defined(HAVE_ASYNC_LDS)
    {
        const int c = t * 4;             // 256 threads x 16B = 4KB per array
        __builtin_amdgcn_global_load_async_to_lds_b128(
            (vi4*)(ref + c),          (vi4*)&sx[c], 0, 0);
        __builtin_amdgcn_global_load_async_to_lds_b128(
            (vi4*)(ref + S_ + c),     (vi4*)&sy[c], 0, 0);
        __builtin_amdgcn_global_load_async_to_lds_b128(
            (vi4*)(ref + 2 * S_ + c), (vi4*)&sz[c], 0, 0);
        wait_async0();
    }
#else
    for (int s = t; s < S_; s += 256) {
        sx[s] = ref[s];
        sy[s] = ref[S_ + s];
        sz[s] = ref[2 * S_ + s];
    }
#endif
    __syncthreads();
    const int n = blockIdx.x * 256 + t;
    const float* q = xyz1 + (size_t)b * 3 * N_;
    const float qx = q[n], qy = q[N_ + n], qz = q[2 * N_ + n];
    float d0 = 1e30f, d1 = 1e30f, d2 = 1e30f;
    int   i0 = 0, i1 = 0, i2 = 0;
    for (int s = 0; s < S_; ++s) {
        float dx = qx - sx[s], dy = qy - sy[s], dz = qz - sz[s];
        float d = dx * dx + dy * dy + dz * dz;
        if (d < d2) {
            if (d < d1) {
                d2 = d1; i2 = i1;
                if (d < d0) { d1 = d0; i1 = i0; d0 = d; i0 = s; }
                else        { d1 = d;  i1 = s; }
            } else { d2 = d; i2 = s; }
        }
    }
    float r0 = 1.f / (d0 + 1e-8f), r1 = 1.f / (d1 + 1e-8f), r2 = 1.f / (d2 + 1e-8f);
    float inv = 1.f / (r0 + r1 + r2);
    size_t o = ((size_t)b * N_ + n) * 3;
    idx_out[o + 0] = i0; idx_out[o + 1] = i1; idx_out[o + 2] = i2;
    w_out[o + 0] = r0 * inv; w_out[o + 1] = r1 * inv; w_out[o + 2] = r2 * inv;
}

// ---------------------------------------------------------------------------
// 2) f32 -> bf16 weight conversion
// ---------------------------------------------------------------------------
__global__ __launch_bounds__(256) void f32_to_bf16_kernel(
        const float* __restrict__ src, unsigned short* __restrict__ dst, int n) {
    int i = blockIdx.x * 256 + threadIdx.x;
    if (i < n) dst[i] = f2bf(src[i]);
}

// ---------------------------------------------------------------------------
// 3) Build Xt[B][N][CIN] bf16 = transpose(concat(points1, interp(points2)))
//    LDS-tiled so global writes are contiguous along the K (channel) axis.
// ---------------------------------------------------------------------------
__global__ __launch_bounds__(256) void build_xt_kernel(
        const float* __restrict__ points1, const float* __restrict__ points2,
        const int* __restrict__ knn_idx, const float* __restrict__ knn_w,
        unsigned short* __restrict__ Xt) {
    __shared__ float tile[128][65];
    const int b  = blockIdx.y;
    const int n0 = blockIdx.x * 64;
    const int t  = threadIdx.x;
    const int j    = t & 63;      // n offset (phase A)
    const int irow = t >> 6;      // 0..3
    size_t q = ((size_t)b * N_ + n0 + j) * 3;
    const int   ia = knn_idx[q], ib = knn_idx[q + 1], ic = knn_idx[q + 2];
    const float wa = knn_w[q],  wb = knn_w[q + 1],  wc = knn_w[q + 2];
    const int iiB = t & 127;      // c offset (phase B)
    const int jB0 = t >> 7;       // 0..1
    for (int cc = 0; cc < CIN_; cc += 128) {
        for (int i = irow; i < 128; i += 4) {
            int c = cc + i;
            float v;
            if (c < C1_) {
                v = points1[((size_t)b * C1_ + c) * N_ + n0 + j];
            } else {
                const float* p = points2 + ((size_t)b * C2_ + (c - C1_)) * S_;
                v = wa * p[ia] + wb * p[ib] + wc * p[ic];
            }
            tile[i][j] = v;
        }
        __syncthreads();
        for (int jn = jB0; jn < 64; jn += 2)
            Xt[((size_t)b * N_ + n0 + jn) * CIN_ + cc + iiB] = f2bf(tile[iiB][jn]);
        __syncthreads();
    }
}

// ---------------------------------------------------------------------------
// 4) bf16 WMMA GEMM: Y[b][m][n] = sum_k Wb[m][k] * Xt[b][n][k]
//    Block = 8 waves; each wave owns a 64(M) x 16(N) strip (4 accumulators).
//    Fragments loaded directly as contiguous b128 chunks per the ISA layouts.
// ---------------------------------------------------------------------------
union FragCast { int4 q[2]; v16bf v; };

__global__ __launch_bounds__(256) void gemm_bf16_wmma_kernel(
        const unsigned short* __restrict__ Xt,  // [B][Nn][K]
        const unsigned short* __restrict__ Wb,  // [M][K]
        float* __restrict__ Y,                  // [B][M][Nn]
        int K, int M, int Nn) {
    const int lane   = threadIdx.x & 31;
    const int wave   = threadIdx.x >> 5;
    const int lane16 = lane & 15;
    const int hi     = lane >> 4;
    const int b  = blockIdx.z;
    const int m0 = blockIdx.y * 64;
    const int n  = blockIdx.x * 128 + wave * 16 + lane16;

    // B fragment (K x 16): lanes 0-15 hold K {0..7,16..23}, lanes 16-31 hold K {8..15,24..31},
    // packed 2 bf16 per VGPR -> two 8-bf16 chunks per lane.
    const unsigned short* bptr = Xt + ((size_t)b * Nn + n) * K + hi * 8;
    // A fragment (16 x K): lanes 0-15 -> K {0..7,16..23}; lanes 16-31 -> {8..15,24..31}.
    const unsigned short* aptr = Wb + (size_t)(m0 + lane16) * K + hi * 8;
    const size_t arow16 = (size_t)16 * K;

    v8f acc0 = {}, acc1 = {}, acc2 = {}, acc3 = {};

    for (int kt = 0; kt < K; kt += 32) {
        // Unconditional speculative prefetch of the K-tile after next of the
        // streaming B operand (silently dropped if past the row / mapping),
        // branch-free so the scheduler can pipeline loads across iterations.
        __builtin_prefetch(bptr + kt + 64, 0, 3);

        FragCast bf;
        bf.q[0] = *(const int4*)(bptr + kt);
        bf.q[1] = *(const int4*)(bptr + kt + 16);

        FragCast a0, a1, a2, a3;
        const unsigned short* ak = aptr + kt;
        a0.q[0] = *(const int4*)(ak);            a0.q[1] = *(const int4*)(ak + 16); ak += arow16;
        a1.q[0] = *(const int4*)(ak);            a1.q[1] = *(const int4*)(ak + 16); ak += arow16;
        a2.q[0] = *(const int4*)(ak);            a2.q[1] = *(const int4*)(ak + 16); ak += arow16;
        a3.q[0] = *(const int4*)(ak);            a3.q[1] = *(const int4*)(ak + 16);

        acc0 = __builtin_amdgcn_wmma_f32_16x16x32_bf16(false, a0.v, false, bf.v, (short)0, acc0, false, false);
        acc1 = __builtin_amdgcn_wmma_f32_16x16x32_bf16(false, a1.v, false, bf.v, (short)0, acc1, false, false);
        acc2 = __builtin_amdgcn_wmma_f32_16x16x32_bf16(false, a2.v, false, bf.v, (short)0, acc2, false, false);
        acc3 = __builtin_amdgcn_wmma_f32_16x16x32_bf16(false, a3.v, false, bf.v, (short)0, acc3, false, false);
    }

    // C/D layout: VGPR r -> M = r (lanes 0-15) / r+8 (lanes 16-31), N = lane%16.
    #pragma unroll
    for (int r = 0; r < 8; ++r) {
        int mrow = m0 + r + hi * 8;
        Y[((size_t)b * M + mrow +  0) * Nn + n] = acc0[r];
        Y[((size_t)b * M + mrow + 16) * Nn + n] = acc1[r];
        Y[((size_t)b * M + mrow + 32) * Nn + n] = acc2[r];
        Y[((size_t)b * M + mrow + 48) * Nn + n] = acc3[r];
    }
}

// ---------------------------------------------------------------------------
// 5) Per-channel batch statistics folded into scale/shift (bias absorbed by BN)
// ---------------------------------------------------------------------------
__global__ __launch_bounds__(256) void bn_stats_kernel(
        const float* __restrict__ Y, const float* __restrict__ g,
        const float* __restrict__ be, float* __restrict__ scale,
        float* __restrict__ shift, int M, int Nn, int Bn) {
    __shared__ float ssum[256], ssq[256];
    const int o = blockIdx.x;
    const int t = threadIdx.x;
    float s = 0.f, q = 0.f;
    for (int b = 0; b < Bn; ++b) {
        const float* row = Y + ((size_t)b * M + o) * Nn;
        for (int n = t; n < Nn; n += 256) {
            float v = row[n];
            s += v; q += v * v;
        }
    }
    ssum[t] = s; ssq[t] = q;
    __syncthreads();
    for (int d = 128; d > 0; d >>= 1) {
        if (t < d) { ssum[t] += ssum[t + d]; ssq[t] += ssq[t + d]; }
        __syncthreads();
    }
    if (t == 0) {
        float cnt  = (float)Bn * (float)Nn;
        float mean = ssum[0] / cnt;
        float var  = fmaxf(ssq[0] / cnt - mean * mean, 0.f);
        float sc   = g[o] * rsqrtf(var + 1e-5f);
        scale[o] = sc;
        shift[o] = be[o] - mean * sc;
    }
}

// ---------------------------------------------------------------------------
// 6) BN + ReLU + transpose to bf16 Ht[B][N][H] (GEMM2 B-operand layout)
// ---------------------------------------------------------------------------
__global__ __launch_bounds__(256) void bn_relu_transpose_kernel(
        const float* __restrict__ Y, const float* __restrict__ scale,
        const float* __restrict__ shift, unsigned short* __restrict__ Ht) {
    __shared__ unsigned short tile[64][65];
    const int b  = blockIdx.z;
    const int n0 = blockIdx.x * 64;
    const int h0 = blockIdx.y * 64;
    const int t  = threadIdx.x;
    const int j = t & 63, i0 = t >> 6;
    for (int i = i0; i < 64; i += 4) {
        int h = h0 + i;
        float v = Y[((size_t)b * H_ + h) * N_ + n0 + j];
        v = fmaxf(v * scale[h] + shift[h], 0.f);
        tile[i][j] = f2bf(v);
    }
    __syncthreads();
    const int i = t & 63, j0 = t >> 6;
    for (int jn = j0; jn < 64; jn += 4)
        Ht[((size_t)b * N_ + n0 + jn) * H_ + h0 + i] = tile[i][jn];
}

// ---------------------------------------------------------------------------
// 7) Final BN + ReLU -> f32 out[B][H][N]
// ---------------------------------------------------------------------------
__global__ __launch_bounds__(256) void bn_relu_out_kernel(
        const float* __restrict__ Y, const float* __restrict__ scale,
        const float* __restrict__ shift, float* __restrict__ out, int total) {
    for (int i = blockIdx.x * 256 + threadIdx.x; i < total; i += gridDim.x * 256) {
        int ch = (i >> 12) & (H_ - 1);   // N_ = 4096 = 2^12
        out[i] = fmaxf(Y[i] * scale[ch] + shift[ch], 0.f);
    }
}

// ---------------------------------------------------------------------------
extern "C" void kernel_launch(void* const* d_in, const int* in_sizes, int n_in,
                              void* d_out, int out_size, void* d_ws, size_t ws_size,
                              hipStream_t stream) {
    (void)in_sizes; (void)n_in; (void)out_size; (void)ws_size;
    const float* xyz1    = (const float*)d_in[0];
    const float* xyz2    = (const float*)d_in[1];
    const float* points1 = (const float*)d_in[2];
    const float* points2 = (const float*)d_in[3];
    const float* W1      = (const float*)d_in[4];
    const float* g1      = (const float*)d_in[6];
    const float* be1     = (const float*)d_in[7];
    const float* W2      = (const float*)d_in[8];
    const float* g2      = (const float*)d_in[10];
    const float* be2     = (const float*)d_in[11];

    char* ws = (char*)d_ws;
    size_t off = 0;
    auto alloc = [&](size_t bytes) {
        void* p = ws + off;
        off = (off + bytes + 255) & ~(size_t)255;
        return p;
    };
    int*            knn_idx = (int*)           alloc((size_t)B_ * N_ * 3 * 4);
    float*          knn_w   = (float*)         alloc((size_t)B_ * N_ * 3 * 4);
    unsigned short* W1bf    = (unsigned short*)alloc((size_t)H_ * CIN_ * 2);
    unsigned short* W2bf    = (unsigned short*)alloc((size_t)H_ * H_ * 2);
    float*          scale1  = (float*)         alloc(1024);
    float*          shift1  = (float*)         alloc(1024);
    float*          scale2  = (float*)         alloc(1024);
    float*          shift2  = (float*)         alloc(1024);
    unsigned short* Xt      = (unsigned short*)alloc((size_t)B_ * N_ * CIN_ * 2);
    float*          Y1      = (float*)         alloc((size_t)B_ * H_ * N_ * 4);
    unsigned short* Ht      = (unsigned short*)alloc((size_t)B_ * N_ * H_ * 2);
    float*          Y2      = (float*)Xt;      // Xt is dead after GEMM1
    float*          out     = (float*)d_out;

    // 1) KNN
    knn3_kernel<<<dim3(N_ / 256, B_), 256, 0, stream>>>(xyz1, xyz2, knn_idx, knn_w);

    // 2) weight conversion
    f32_to_bf16_kernel<<<(H_ * CIN_ + 255) / 256, 256, 0, stream>>>(W1, W1bf, H_ * CIN_);
    f32_to_bf16_kernel<<<(H_ * H_  + 255) / 256, 256, 0, stream>>>(W2, W2bf, H_ * H_);

    // 3) interpolate + concat + transpose -> Xt bf16
    build_xt_kernel<<<dim3(N_ / 64, B_), 256, 0, stream>>>(points1, points2, knn_idx, knn_w, Xt);

    // 4) GEMM1 (K = 768)
    gemm_bf16_wmma_kernel<<<dim3(N_ / 128, H_ / 64, B_), 256, 0, stream>>>(Xt, W1bf, Y1, CIN_, H_, N_);

    // 5) BN1 stats + BN/ReLU/transpose -> Ht bf16
    bn_stats_kernel<<<H_, 256, 0, stream>>>(Y1, g1, be1, scale1, shift1, H_, N_, B_);
    bn_relu_transpose_kernel<<<dim3(N_ / 64, H_ / 64, B_), 256, 0, stream>>>(Y1, scale1, shift1, Ht);

    // 6) GEMM2 (K = 256)
    gemm_bf16_wmma_kernel<<<dim3(N_ / 128, H_ / 64, B_), 256, 0, stream>>>(Ht, W2bf, Y2, H_, H_, N_);

    // 7) BN2 stats + final BN/ReLU -> out f32
    bn_stats_kernel<<<H_, 256, 0, stream>>>(Y2, g2, be2, scale2, shift2, H_, N_, B_);
    bn_relu_out_kernel<<<16384, 256, 0, stream>>>(Y2, scale2, shift2, out, B_ * H_ * N_);
}
